// Attention_17042430231279
// MI455X (gfx1250) — compile-verified, matrix operand order checked by reading, not run on hardware
//
#include <hip/hip_runtime.h>
#include <hip/hip_bf16.h>
#include <cstdint>
#include <cstddef>

typedef __attribute__((ext_vector_type(16))) _Float16 v16h;
typedef __attribute__((ext_vector_type(8)))  _Float16 v8h;
typedef __attribute__((ext_vector_type(8)))  float    v8f;

#define NPIX  1600   // 40*40
#define HDIM  40
#define CDIM  128
#define NHEADS 8
#define CHPAD 32     // per-head channels padded 16->32 for K=32 WMMA

// ---------------------------------------------------------------------------
// Load 16 f16 A/B-operand elements for one lane of v_wmma_f32_16x16x32_f16.
// Lane layout (ISA 7.12.2, 16-bit A 16x32): lane<16 -> K {0..7, 16..23},
// lane>=16 -> K {8..15, 24..31}. Caller bakes the (lane>>4)*8 offset into p,
// so elements 0..7 are p[0..7] and elements 8..15 are p[16..23]: two 16B loads.
// ---------------------------------------------------------------------------
__device__ __forceinline__ v16h load_op16(const _Float16* p) {
    v8h lo = *(const v8h*)(p);
    v8h hi = *(const v8h*)(p + 16);
    v16h r;
#pragma unroll
    for (int i = 0; i < 8; ++i) { r[i] = lo[i]; r[8 + i] = hi[i]; }
    return r;
}

// ---------------------------------------------------------------------------
// Batched, register-blocked WMMA GEMM:
//   C[b] = A[b] (MxK, row-major) * Bt[b]^T   (Bt stored NxK, row-major)
// f16 operands, f32 accumulate. Each wave owns a (16*BM)x(16*BN) C region
// (BMxBN accumulators), so per K-chunk it issues BM+BN fragment loads for
// BM*BN wmma ops. K multiple of 32; M % (16*BM) == 0; N % (16*BN) == 0.
// ---------------------------------------------------------------------------
template<int BM, int BN>
__global__ __launch_bounds__(256) void wmma_gemm_kernel(
    const _Float16* __restrict__ A, const _Float16* __restrict__ Bt,
    float* __restrict__ C, int M, int N, int K,
    long long sA, long long sB, long long sC)
{
    const int wave = threadIdx.x >> 5;
    const int lane = threadIdx.x & 31;
    const int tilesN = N / (16 * BN);
    const int tiles  = (M / (16 * BM)) * tilesN;
    const int t = blockIdx.x * 8 + wave;
    if (t >= tiles) return;                       // wave-uniform: EXEC stays all-1s
    const int tm = t / tilesN;
    const int tn = t - tm * tilesN;

    const _Float16* Ap[BM];
    const _Float16* Bp[BN];
#pragma unroll
    for (int i = 0; i < BM; ++i)
        Ap[i] = A + (long long)blockIdx.z * sA
              + (size_t)(tm * 16 * BM + i * 16 + (lane & 15)) * (size_t)K
              + (size_t)((lane >> 4) * 8);
#pragma unroll
    for (int j = 0; j < BN; ++j)
        Bp[j] = Bt + (long long)blockIdx.z * sB
              + (size_t)(tn * 16 * BN + j * 16 + (lane & 15)) * (size_t)K
              + (size_t)((lane >> 4) * 8);

    v8f acc[BM][BN] = {};
    for (int k0 = 0; k0 < K; k0 += 32) {
        // prefetch the streamed A operand a few chunks ahead (L2-resident data;
        // lowers to global_prefetch_b8, speculative so tail over-run is safe)
        __builtin_prefetch(Ap[0] + k0 + 512, 0, 0);
        v16h a[BM], b[BN];
#pragma unroll
        for (int i = 0; i < BM; ++i) a[i] = load_op16(Ap[i] + k0);
#pragma unroll
        for (int j = 0; j < BN; ++j) b[j] = load_op16(Bp[j] + k0);
#pragma unroll
        for (int i = 0; i < BM; ++i)
#pragma unroll
            for (int j = 0; j < BN; ++j)
                acc[i][j] = __builtin_amdgcn_wmma_f32_16x16x32_f16(
                    /*neg_a=*/false, a[i], /*neg_b=*/false, b[j],
                    /*c_mod=*/(short)0, acc[i][j],
                    /*reuse_a=*/false, /*reuse_b=*/false);
    }

    // C/D layout: VGPR r -> (lane<16: M=r, N=lane) (lane>=16: M=8+r, N=lane-16)
#pragma unroll
    for (int i = 0; i < BM; ++i) {
        const int row = tm * 16 * BM + i * 16 + ((lane >> 4) << 3);
#pragma unroll
        for (int j = 0; j < BN; ++j) {
            const int col = tn * 16 * BN + j * 16 + (lane & 15);
            float* Cp = C + (long long)blockIdx.z * sC + (size_t)row * (size_t)N + col;
#pragma unroll
            for (int r = 0; r < 8; ++r) Cp[(size_t)r * N] = acc[i][j][r];
        }
    }
}

// ---------------------------------------------------------------------------
// Small utility kernels
// ---------------------------------------------------------------------------
__global__ void cvt_f32_to_f16_kernel(const float* __restrict__ s, _Float16* __restrict__ d, int n) {
    int i = blockIdx.x * 256 + threadIdx.x;
    if (i < n) d[i] = (_Float16)s[i];
}

// src is (CDIM x NPIX) channel-major f32 -> dst (NPIX x CDIM) pixel-major f16
__global__ void transpose_cvt_kernel(const float* __restrict__ s, _Float16* __restrict__ d) {
    int i = blockIdx.x * 256 + threadIdx.x;
    if (i >= CDIM * NPIX) return;
    int c = i / NPIX, n = i - c * NPIX;
    d[(size_t)n * CDIM + c] = (_Float16)s[i];
}

__global__ void fill_f16_kernel(_Float16* __restrict__ d, int n) {
    int i = blockIdx.x * 256 + threadIdx.x;
    if (i < n) d[i] = (_Float16)0.0f;
}

// ---------------------------------------------------------------------------
// pe bias + LayerNorm(channels) + residual; emits xr (f32, c-major for convs)
// and xrh (f16, n-major for the kv GEMM Bt operand). One block per pixel.
// ---------------------------------------------------------------------------
__global__ __launch_bounds__(128) void ln_residual_kernel(
    const float* __restrict__ pe, const float* __restrict__ pe_b,
    const float* __restrict__ g, const float* __restrict__ b,
    const float* __restrict__ x, float* __restrict__ xr,
    _Float16* __restrict__ xrh)
{
    __shared__ float red[128];
    const int c = threadIdx.x, n = blockIdx.x;
    float t = pe[(size_t)c * NPIX + n] + pe_b[c];
    red[c] = t; __syncthreads();
    for (int off = 64; off > 0; off >>= 1) { if (c < off) red[c] += red[c + off]; __syncthreads(); }
    const float mu = red[0] * (1.0f / 128.0f);
    __syncthreads();
    const float dv = t - mu;
    red[c] = dv * dv; __syncthreads();
    for (int off = 64; off > 0; off >>= 1) { if (c < off) red[c] += red[c + off]; __syncthreads(); }
    const float var = red[0] * (1.0f / 128.0f);
    const float y = dv * rsqrtf(var + 1e-5f) * g[c] + b[c];
    const float xv = x[(size_t)c * NPIX + n] + y;
    xr[(size_t)c * NPIX + n] = xv;
    xrh[(size_t)n * CDIM + c] = (_Float16)xv;
}

// ---------------------------------------------------------------------------
// Direct dilated 3x3 conv (pad=dil) + BN + ReLU, writes f16 into the concat
// buffer (pixel-major, 256 channels) at channel offset catOff.
// ---------------------------------------------------------------------------
__global__ __launch_bounds__(256) void conv3x3_bn_relu_kernel(
    const float* __restrict__ xr, const float* __restrict__ w,
    const float* __restrict__ g, const float* __restrict__ bb,
    const float* __restrict__ m, const float* __restrict__ v,
    _Float16* __restrict__ cat, int dil, int catOff)
{
    int idx = blockIdx.x * 256 + threadIdx.x;
    if (idx >= CDIM * NPIX) return;
    const int o = idx / NPIX, n = idx - o * NPIX;
    const int y = n / HDIM, xx = n - y * HDIM;
    float acc = 0.0f;
    for (int ci = 0; ci < CDIM; ++ci) {
        const float* xc = xr + (size_t)ci * NPIX;
        const float* wc = w + ((size_t)o * CDIM + ci) * 9;
#pragma unroll
        for (int ty = 0; ty < 3; ++ty) {
            const int yy = y + (ty - 1) * dil;
            if (yy < 0 || yy >= HDIM) continue;
#pragma unroll
            for (int tx = 0; tx < 3; ++tx) {
                const int xc2 = xx + (tx - 1) * dil;
                if (xc2 < 0 || xc2 >= HDIM) continue;
                acc += xc[yy * HDIM + xc2] * wc[ty * 3 + tx];
            }
        }
    }
    const float s = g[o] * rsqrtf(v[o] + 1e-5f);
    float val = (acc - m[o]) * s + bb[o];
    val = fmaxf(val, 0.0f);
    cat[(size_t)n * 256 + catOff + o] = (_Float16)val;
}

__global__ void bnrelu_kernel(float* __restrict__ q,
                              const float* __restrict__ g, const float* __restrict__ b,
                              const float* __restrict__ m, const float* __restrict__ v)
{
    int idx = blockIdx.x * 256 + threadIdx.x;
    if (idx >= CDIM * NPIX) return;
    const int o = idx / NPIX;
    const float s = g[o] * rsqrtf(v[o] + 1e-5f);
    q[idx] = fmaxf((q[idx] - m[o]) * s + b[o], 0.0f);
}

// depthwise 3x3 pad 1 over 256 channels
__global__ void dwconv_kernel(const float* __restrict__ kv, const float* __restrict__ w,
                              float* __restrict__ out)
{
    int idx = blockIdx.x * 256 + threadIdx.x;
    if (idx >= 256 * NPIX) return;
    const int c = idx / NPIX, n = idx - c * NPIX;
    const int y = n / HDIM, x = n - y * HDIM;
    const float* wc = w + (size_t)c * 9;
    const float* src = kv + (size_t)c * NPIX;
    float acc = 0.0f;
#pragma unroll
    for (int ty = 0; ty < 3; ++ty) {
        const int yy = y + ty - 1;
        if (yy < 0 || yy >= HDIM) continue;
#pragma unroll
        for (int tx = 0; tx < 3; ++tx) {
            const int xx = x + tx - 1;
            if (xx < 0 || xx >= HDIM) continue;
            acc += src[yy * HDIM + xx] * wc[ty * 3 + tx];
        }
    }
    out[idx] = acc;
}

// L2-normalize row o (length NPIX) of src (128 x NPIX) and pack into per-head
// padded f16 operand dst[head][n*CHPAD + c] (pad channels pre-zeroed).
__global__ __launch_bounds__(256) void rownorm_pack_kernel(
    const float* __restrict__ src, _Float16* __restrict__ dst)
{
    __shared__ float red[256];
    const int o = blockIdx.x, tid = threadIdx.x;
    const float* row = src + (size_t)o * NPIX;
    float p = 0.0f;
    for (int n = tid; n < NPIX; n += 256) { const float v = row[n]; p += v * v; }
    red[tid] = p; __syncthreads();
    for (int off = 128; off > 0; off >>= 1) { if (tid < off) red[tid] += red[tid + off]; __syncthreads(); }
    const float inv = 1.0f / fmaxf(sqrtf(red[0]), 1e-12f);
    const int h = o >> 4, c = o & 15;
    _Float16* d = dst + (size_t)h * NPIX * CHPAD + c;
    for (int n = tid; n < NPIX; n += 256) d[(size_t)n * CHPAD] = (_Float16)(row[n] * inv);
}

// vh[h][c*NPIX + m] = kvd[(128 + h*16 + c)*NPIX + m]
__global__ void vpack_kernel(const float* __restrict__ kvd, _Float16* __restrict__ vh) {
    int idx = blockIdx.x * 256 + threadIdx.x;
    if (idx >= NHEADS * 16 * NPIX) return;
    vh[idx] = (_Float16)kvd[(size_t)CDIM * NPIX + idx];   // heads/channels are contiguous
}

// oh[n*128 + h*16+c] = of[h][n*16 + c]  (attention output -> GEMM Bt operand)
__global__ void opack_kernel(const float* __restrict__ of, _Float16* __restrict__ oh) {
    int idx = blockIdx.x * 256 + threadIdx.x;
    if (idx >= NPIX * CDIM) return;
    const int n = idx / CDIM, cc = idx - n * CDIM;
    const int h = cc >> 4, c = cc & 15;
    oh[idx] = (_Float16)of[((size_t)h * NPIX + n) * 16 + c];
}

// ---------------------------------------------------------------------------
// Fused quad top-k softmax. One block per (head,row): bitonic-sort a 2048-pad
// copy of the scaled row in LDS to get all 4 thresholds (k-1 indices of the
// descending order), then combine the four masked softmaxes into one f16
// weight row W = sum_i a_i * softmax_i.
// ---------------------------------------------------------------------------
__global__ __launch_bounds__(256) void softmax_topk_kernel(
    const float* __restrict__ attn, _Float16* __restrict__ W,
    const float* __restrict__ temperature,
    const float* __restrict__ a1, const float* __restrict__ a2,
    const float* __restrict__ a3, const float* __restrict__ a4)
{
    __shared__ float s[2048];
    const int tid = threadIdx.x;
    const int row = blockIdx.x, head = blockIdx.y;
    const float temp = temperature[head];
    const float* arow = attn + ((long long)head * NPIX + row) * NPIX;

    for (int i = tid; i < 2048; i += 256)
        s[i] = (i < NPIX) ? arow[i] * temp : -3.4e38f;
    __syncthreads();

    // bitonic sort, descending
    for (int k2 = 2; k2 <= 2048; k2 <<= 1) {
        for (int j = k2 >> 1; j > 0; j >>= 1) {
            for (int i = tid; i < 2048; i += 256) {
                const int p = i ^ j;
                if (p > i) {
                    const float a = s[i], b = s[p];
                    const bool desc = ((i & k2) == 0);
                    if (desc ? (a < b) : (a > b)) { s[i] = b; s[p] = a; }
                }
            }
            __syncthreads();
        }
    }

    const float maxv = s[0];
    const float t0 = s[799];    // k = N/2   = 800
    const float t1 = s[1065];   // k = 2N/3  = 1066
    const float t2 = s[1199];   // k = 3N/4  = 1200
    const float t3 = s[1279];   // k = 4N/5  = 1280
    __syncthreads();

    float p0 = 0, p1 = 0, p2 = 0, p3 = 0;
    for (int m = tid; m < NPIX; m += 256) {
        const float a = arow[m] * temp;
        const float e = __expf(a - maxv);
        if (a >= t0) p0 += e;
        if (a >= t1) p1 += e;
        if (a >= t2) p2 += e;
        if (a >= t3) p3 += e;
    }
    s[tid] = p0; s[256 + tid] = p1; s[512 + tid] = p2; s[768 + tid] = p3;
    __syncthreads();
    for (int off = 128; off > 0; off >>= 1) {
        if (tid < off) {
            s[tid]       += s[tid + off];
            s[256 + tid] += s[256 + tid + off];
            s[512 + tid] += s[512 + tid + off];
            s[768 + tid] += s[768 + tid + off];
        }
        __syncthreads();
    }
    const float c0 = a1[0] / s[0],   c1 = a2[0] / s[256];
    const float c2 = a3[0] / s[512], c3 = a4[0] / s[768];

    _Float16* wrow = W + ((long long)head * NPIX + row) * NPIX;
    for (int m = tid; m < NPIX; m += 256) {
        const float a = arow[m] * temp;
        const float e = __expf(a - maxv);
        float wv = 0.0f;
        if (a >= t0) wv += c0 * e;
        if (a >= t1) wv += c1 * e;
        if (a >= t2) wv += c2 * e;
        if (a >= t3) wv += c3 * e;
        wrow[m] = (_Float16)wv;
    }
}

// ---------------------------------------------------------------------------
// Host orchestration
// ---------------------------------------------------------------------------
extern "C" void kernel_launch(void* const* d_in, const int* in_sizes, int n_in,
                              void* d_out, int out_size, void* d_ws, size_t ws_size,
                              hipStream_t stream) {
    (void)in_sizes; (void)n_in; (void)out_size; (void)ws_size;

    const float* x       = (const float*)d_in[0];
    const float* pe_w    = (const float*)d_in[1];
    const float* pe_b    = (const float*)d_in[2];
    const float* ln_g    = (const float*)d_in[3];
    const float* ln_b    = (const float*)d_in[4];
    const float* aspp_w1 = (const float*)d_in[5];
    const float* bn1_g   = (const float*)d_in[6];
    const float* bn1_b   = (const float*)d_in[7];
    const float* bn1_m   = (const float*)d_in[8];
    const float* bn1_v   = (const float*)d_in[9];
    const float* aspp_w2 = (const float*)d_in[10];
    const float* bn2_g   = (const float*)d_in[11];
    const float* bn2_b   = (const float*)d_in[12];
    const float* bn2_m   = (const float*)d_in[13];
    const float* bn2_v   = (const float*)d_in[14];
    const float* proj_w  = (const float*)d_in[15];
    const float* bnp_g   = (const float*)d_in[16];
    const float* bnp_b   = (const float*)d_in[17];
    const float* bnp_m   = (const float*)d_in[18];
    const float* bnp_v   = (const float*)d_in[19];
    const float* kv_w    = (const float*)d_in[20];
    const float* kvdw_w  = (const float*)d_in[21];
    const float* out_w   = (const float*)d_in[22];
    const float* temper  = (const float*)d_in[23];
    const float* a1      = (const float*)d_in[24];
    const float* a2      = (const float*)d_in[25];
    const float* a3      = (const float*)d_in[26];
    const float* a4      = (const float*)d_in[27];

    char* wsp = (char*)d_ws;
    size_t off = 0;
    auto carve = [&](size_t bytes) -> void* {
        void* p = wsp + off;
        off = (off + bytes + 255) & ~(size_t)255;
        return p;
    };

    _Float16* xh     = (_Float16*)carve((size_t)NPIX * CDIM * 2);     // x, pixel-major f16
    _Float16* wpeh   = (_Float16*)carve((size_t)CDIM * CDIM * 2);
    float*    pe     = (float*)   carve((size_t)CDIM * NPIX * 4);
    float*    xr     = (float*)   carve((size_t)CDIM * NPIX * 4);     // residual, channel-major
    _Float16* xrh    = (_Float16*)carve((size_t)NPIX * CDIM * 2);     // residual, pixel-major f16
    _Float16* cat    = (_Float16*)carve((size_t)NPIX * 256 * 2);      // [b1;b2], pixel-major f16
    _Float16* projwh = (_Float16*)carve((size_t)CDIM * 256 * 2);
    float*    qf     = (float*)   carve((size_t)CDIM * NPIX * 4);
    _Float16* kvwh   = (_Float16*)carve((size_t)256 * CDIM * 2);
    float*    kvf    = (float*)   carve((size_t)256 * NPIX * 4);
    float*    kvd    = (float*)   carve((size_t)256 * NPIX * 4);
    _Float16* qh     = (_Float16*)carve((size_t)NHEADS * NPIX * CHPAD * 2);
    _Float16* kh     = (_Float16*)carve((size_t)NHEADS * NPIX * CHPAD * 2);
    _Float16* vh     = (_Float16*)carve((size_t)NHEADS * 16 * NPIX * 2);
    float*    attn   = (float*)   carve((size_t)NHEADS * NPIX * NPIX * 4);
    _Float16* Wh     = (_Float16*)carve((size_t)NHEADS * NPIX * NPIX * 2);
    float*    of     = (float*)   carve((size_t)NHEADS * NPIX * 16 * 4);
    _Float16* oh     = (_Float16*)carve((size_t)NPIX * CDIM * 2);
    _Float16* outwh  = (_Float16*)carve((size_t)CDIM * CDIM * 2);

    // 32x32-per-wave blocked GEMM (BM=2, BN=2): M%32==0, N%32==0
    auto gemm22 = [&](const _Float16* A, const _Float16* Bt, float* C,
                      int M, int N, int K, long long sA, long long sB, long long sC, int batch) {
        const int tiles = (M / 32) * (N / 32);
        dim3 grid((tiles + 7) / 8, 1, batch);
        wmma_gemm_kernel<2, 2><<<grid, 256, 0, stream>>>(A, Bt, C, M, N, K, sA, sB, sC);
    };
    // 32x16-per-wave blocked GEMM (BM=2, BN=1): for the N=16 attn*V GEMM
    auto gemm21 = [&](const _Float16* A, const _Float16* Bt, float* C,
                      int M, int N, int K, long long sA, long long sB, long long sC, int batch) {
        const int tiles = (M / 32) * (N / 16);
        dim3 grid((tiles + 7) / 8, 1, batch);
        wmma_gemm_kernel<2, 1><<<grid, 256, 0, stream>>>(A, Bt, C, M, N, K, sA, sB, sC);
    };
    auto cvt = [&](const float* s, _Float16* d, int n) {
        cvt_f32_to_f16_kernel<<<(n + 255) / 256, 256, 0, stream>>>(s, d, n);
    };

    // --- weight conversions + x transpose-pack ---
    cvt(pe_w,   wpeh,   CDIM * CDIM);
    cvt(proj_w, projwh, CDIM * 256);
    cvt(kv_w,   kvwh,   256 * CDIM);
    cvt(out_w,  outwh,  CDIM * CDIM);
    transpose_cvt_kernel<<<(CDIM * NPIX + 255) / 256, 256, 0, stream>>>(x, xh);

    // --- pe = pe_w * x (1x1 conv as GEMM) ; LN + residual ---
    gemm22(wpeh, xh, pe, CDIM, NPIX, CDIM, 0, 0, 0, 1);
    ln_residual_kernel<<<NPIX, 128, 0, stream>>>(pe, pe_b, ln_g, ln_b, x, xr, xrh);

    // --- dilated ASPP branches -> concat (f16, pixel-major) ---
    const int convBlocks = (CDIM * NPIX + 255) / 256;
    conv3x3_bn_relu_kernel<<<convBlocks, 256, 0, stream>>>(xr, aspp_w1, bn1_g, bn1_b, bn1_m, bn1_v, cat, 3, 0);
    conv3x3_bn_relu_kernel<<<convBlocks, 256, 0, stream>>>(xr, aspp_w2, bn2_g, bn2_b, bn2_m, bn2_v, cat, 5, CDIM);

    // --- q = relu(bn(proj_w * cat)) ---
    gemm22(projwh, cat, qf, CDIM, NPIX, 256, 0, 0, 0, 1);
    bnrelu_kernel<<<convBlocks, 256, 0, stream>>>(qf, bnp_g, bnp_b, bnp_m, bnp_v);

    // --- kv = depthwise3x3(kv_w * xr) ---
    gemm22(kvwh, xrh, kvf, 256, NPIX, CDIM, 0, 0, 0, 1);
    dwconv_kernel<<<(256 * NPIX + 255) / 256, 256, 0, stream>>>(kvf, kvdw_w, kvd);

    // --- L2-normalize q,k rows; pack into padded per-head f16 operands ---
    const int qkElems = NHEADS * NPIX * CHPAD;
    fill_f16_kernel<<<(qkElems + 255) / 256, 256, 0, stream>>>(qh, qkElems);
    fill_f16_kernel<<<(qkElems + 255) / 256, 256, 0, stream>>>(kh, qkElems);
    rownorm_pack_kernel<<<CDIM, 256, 0, stream>>>(qf, qh);
    rownorm_pack_kernel<<<CDIM, 256, 0, stream>>>(kvd, kh);   // k = first 128 rows of kvd
    vpack_kernel<<<(NHEADS * 16 * NPIX + 255) / 256, 256, 0, stream>>>(kvd, vh);

    // --- attn = q^T k per head (M=N=1600, K=32 padded) ---
    gemm22(qh, kh, attn, NPIX, NPIX, CHPAD,
           (long long)NPIX * CHPAD, (long long)NPIX * CHPAD, (long long)NPIX * NPIX, NHEADS);

    // --- fused 4-way top-k softmax -> combined f16 weights ---
    softmax_topk_kernel<<<dim3(NPIX, NHEADS), 256, 0, stream>>>(attn, Wh, temper, a1, a2, a3, a4);

    // --- out = W * v^T per head (M=1600, N=16, K=1600); W is pure-streamed ---
    gemm21(Wh, vh, of, NPIX, 16, NPIX,
           (long long)NPIX * NPIX, (long long)16 * NPIX, (long long)NPIX * 16, NHEADS);

    // --- repack and final 1x1 conv ---
    opack_kernel<<<(NPIX * CDIM + 255) / 256, 256, 0, stream>>>(of, oh);
    gemm22(outwh, oh, (float*)d_out, CDIM, NPIX, CDIM, 0, 0, 0, 1);
}
